// RxnPredictor_1640677507358
// MI455X (gfx1250) — matrix operand order repaired and checked by reading, not compiled, not run on hardware
//
#include <hip/hip_runtime.h>
#include <hip/hip_bf16.h>

// ---------------------------------------------------------------------------
// Types for WMMA fragments (CDNA5 / gfx1250, wave32)
// ---------------------------------------------------------------------------
typedef __attribute__((ext_vector_type(16))) __bf16        v16bf;
typedef __attribute__((ext_vector_type(8)))  float         v8f;
typedef __attribute__((ext_vector_type(4)))  unsigned int  v4u;

#define HDIM   256
#define MAXNEI 10
#define NTILES 4            // N-tiles (of 16) per block => 64 columns / block
#define KMAX   352          // largest K (Wo concat GEMM)
#define LPAD   8            // LDS row pad (elements) -> conflict-free b128 reads

__device__ __forceinline__ unsigned short f2bf(float f) {
    unsigned int x = __float_as_uint(f);
    unsigned int r = x + 0x7fffu + ((x >> 16) & 1u);   // round-to-nearest-even
    return (unsigned short)(r >> 16);
}
__device__ __forceinline__ float bf2f(unsigned int u) {
    return __uint_as_float(u << 16);
}

union FragB16 { v16bf v; v4u q[2]; };

// ---------------------------------------------------------------------------
// Generic f32->bf16 convert with column padding / placement.
// dst[r*ldd + dcol + c] = (c < copyc) ? bf16(src[r*lds + scol + c]) : 0
// ---------------------------------------------------------------------------
__global__ void k_cvt_pad(const float* __restrict__ src, int lds, int scol,
                          unsigned short* __restrict__ dst, int ldd, int dcol,
                          int rows, int copyc, int totc) {
    long long idx   = (long long)blockIdx.x * blockDim.x + threadIdx.x;
    long long total = (long long)rows * totc;
    long long step  = (long long)gridDim.x * blockDim.x;
    for (; idx < total; idx += step) {
        int r = (int)(idx / totc);
        int c = (int)(idx % totc);
        unsigned short v = 0;
        if (c < copyc) v = f2bf(src[(long long)r * lds + scol + c]);
        dst[(long long)r * ldd + dcol + c] = v;
    }
}

// ---------------------------------------------------------------------------
// WMMA GEMM: C[M,N] = A[M,K](bf16) * W[N,K](bf16)^T  (i.e. x @ W.T)
//
// Block = 256 threads = 8 waves. Block computes a 128(M) x 64(N) C tile:
//  - W panel [64 x K] staged once into LDS (shared by all 8 waves),
//    row stride K+LPAD so 16 lanes hit 16 distinct bank groups (ds_load_b128).
//  - Each wave owns one 16-row M tile and NTILES=4 accumulators; one A
//    fragment load feeds 4 v_wmma_f32_16x16x32_bf16 issues.
//  - All 4 B fragments of a k-step are loaded into distinct registers before
//    the WMMA burst: ds_loads clause together and no B-register recycling
//    (avoids WMMA->VALU WAR hazard NOPs and per-WMMA dscnt-0 waits).
// Epilogue: x = acc + bias[n] + addsrc[m*N+n]
//           outf[m*N+n]   = relu_f ? relu(x) : x     (if outf)
//           outb[m*ldo+n] = bf16(relu(x))            (if outb)
// K multiple of 32; M multiple of 16 (tail waves masked); N multiple of 64.
// ---------------------------------------------------------------------------
__global__ void k_gemm_wmma(const unsigned short* __restrict__ A, int lda,
                            const unsigned short* __restrict__ W, int ldw,
                            int M, int N, int K,
                            const float* __restrict__ addsrc,
                            const float* __restrict__ bias,
                            float* __restrict__ outf, int relu_f,
                            unsigned short* __restrict__ outb, int ldo) {
    __shared__ unsigned short sW[NTILES * 16 * (KMAX + LPAD)];
    const int tid  = threadIdx.x;
    const int lane = tid & 31;
    const int wave = tid >> 5;
    const int n0   = blockIdx.y * (NTILES * 16);
    const int lstr = K + LPAD;                 // LDS row stride (elements)

    // ---- cooperative stage of W panel [64 x K] into LDS (16B per move) ----
    const int vecs = (NTILES * 16 * K) >> 3;   // number of 8-ushort vectors
    for (int i = tid; i < vecs; i += 256) {
        const int e = i << 3;
        const int r = e / K;
        const int c = e - r * K;               // K mult of 32 -> 16B aligned
        *(v4u*)(&sW[r * lstr + c]) = *(const v4u*)(W + (long long)(n0 + r) * ldw + c);
    }
    __syncthreads();

    const int mt = blockIdx.x * 8 + wave;
    const int m0 = mt * 16;
    if (m0 < M) {                              // wave-uniform: EXEC all-ones
        const int half = lane >> 4;            // 0: lanes 0-15, 1: lanes 16-31
        const int l15  = lane & 15;

        // A-fragment rows: M = l15 for both halves (ISA 16-bit A 16x32 layout)
        const unsigned short* arow = A + (long long)(m0 + l15) * lda;
        // B-fragment base row in LDS: N = l15 for both halves
        const unsigned short* bbase = &sW[l15 * lstr + half * 16];

        v8f acc[NTILES];
#pragma unroll
        for (int t = 0; t < NTILES; ++t) acc[t] = (v8f){};

        for (int k0 = 0; k0 < K; k0 += 32) {
            FragB16 a;
            // A lane layout: e=0..15 -> K = k0 + half*16 + e
            a.q[0] = *(const v4u*)(arow + k0 + half * 16);
            a.q[1] = *(const v4u*)(arow + k0 + half * 16 + 8);
            if (k0 + 64 <= K) {
                __builtin_prefetch(arow + k0 + 32, 0, 1);  // global_prefetch_b8
            }
            // Load ALL B fragments first (clause the ds_load_b128s) ...
            FragB16 b[NTILES];
#pragma unroll
            for (int t = 0; t < NTILES; ++t) {
                const unsigned short* brow = bbase + t * 16 * lstr + k0;
                b[t].q[0] = *(const v4u*)(brow);
                b[t].q[1] = *(const v4u*)(brow + 8);
            }
            // ... then issue the WMMA burst.
#pragma unroll
            for (int t = 0; t < NTILES; ++t) {
                acc[t] = __builtin_amdgcn_wmma_f32_16x16x32_bf16(
                             false, a.v, false, b[t].v, (short)0, acc[t],
                             false, false);
            }
        }

        // ---- epilogue ----
#pragma unroll
        for (int t = 0; t < NTILES; ++t) {
            const int   n  = n0 + t * 16 + l15;
            const float bv = bias ? bias[n] : 0.0f;
#pragma unroll
            for (int v = 0; v < 8; ++v) {
                const int m = m0 + half * 8 + v;   // C layout: VGPR v -> M=half*8+v
                float x = acc[t][v] + bv;
                if (addsrc) x += addsrc[(long long)m * N + n];
                if (outf)   outf[(long long)m * N + n] = relu_f ? fmaxf(x, 0.0f) : x;
                if (outb)   outb[(long long)m * ldo + n] = f2bf(fmaxf(x, 0.0f));
            }
        }
    }
}

// ---------------------------------------------------------------------------
// Neighbor gather-sum: out[row, coff + c] = bf16( sum_j msg[graph[row,j], c] )
// One wave per row; each lane owns 8 consecutive columns (H=256).
// ---------------------------------------------------------------------------
__global__ void k_gather_sum(const unsigned short* __restrict__ msg,
                             const int* __restrict__ graph, int M,
                             unsigned short* __restrict__ out, int ldo, int coff) {
    const int lane = threadIdx.x & 31;
    const int row  = (int)(((long long)blockIdx.x * blockDim.x + threadIdx.x) >> 5);
    if (row >= M) return;
    const int c0 = lane * 8;
    float acc[8] = {0.f, 0.f, 0.f, 0.f, 0.f, 0.f, 0.f, 0.f};
#pragma unroll
    for (int j = 0; j < MAXNEI; ++j) {
        const int nb = graph[row * MAXNEI + j];
        v4u d = *(const v4u*)(msg + (long long)nb * HDIM + c0);
        acc[0] += bf2f(d.x & 0xffffu); acc[1] += bf2f(d.x >> 16);
        acc[2] += bf2f(d.y & 0xffffu); acc[3] += bf2f(d.y >> 16);
        acc[4] += bf2f(d.z & 0xffffu); acc[5] += bf2f(d.z >> 16);
        acc[6] += bf2f(d.w & 0xffffu); acc[7] += bf2f(d.w >> 16);
    }
    v4u o;
    o.x = (unsigned int)f2bf(acc[0]) | ((unsigned int)f2bf(acc[1]) << 16);
    o.y = (unsigned int)f2bf(acc[2]) | ((unsigned int)f2bf(acc[3]) << 16);
    o.z = (unsigned int)f2bf(acc[4]) | ((unsigned int)f2bf(acc[5]) << 16);
    o.w = (unsigned int)f2bf(acc[6]) | ((unsigned int)f2bf(acc[7]) << 16);
    *(v4u*)(out + (long long)row * ldo + coff) = o;
}

// ---------------------------------------------------------------------------
// Segment sum over sorted mol_idx (no atomics -> deterministic).
// One block per molecule; 256 threads = 256 columns.
// ---------------------------------------------------------------------------
__global__ void k_segsum(const float* __restrict__ ah,
                         const int* __restrict__ midx, int n_atoms,
                         float* __restrict__ mol) {
    const int m = blockIdx.x;
    int lo = 0, hi = n_atoms;
    while (lo < hi) { int mid = (lo + hi) >> 1; if (midx[mid] < m) lo = mid + 1; else hi = mid; }
    const int start = lo;
    hi = n_atoms;
    while (lo < hi) { int mid = (lo + hi) >> 1; if (midx[mid] < m + 1) lo = mid + 1; else hi = mid; }
    const int end = lo;
    const int c = threadIdx.x;
    float acc = 0.f;
    for (int a = start; a < end; ++a) acc += ah[(long long)a * HDIM + c];
    mol[(long long)m * HDIM + c] = acc;
}

// ---------------------------------------------------------------------------
// diff -> bf16
// ---------------------------------------------------------------------------
__global__ void k_diff_bf(const float* __restrict__ t, const float* __restrict__ s,
                          unsigned short* __restrict__ o, int n) {
    int i = blockIdx.x * blockDim.x + threadIdx.x;
    if (i < n) o[i] = f2bf(t[i] - s[i]);
}

// ---------------------------------------------------------------------------
// Final projection: out[m] = sum_c rxn[m,c]*Wro[c] + bro. One wave per row.
// ---------------------------------------------------------------------------
__global__ void k_final(const unsigned short* __restrict__ rxn,
                        const float* __restrict__ Wro, const float* __restrict__ bro,
                        float* __restrict__ out, int M) {
    const int lane = threadIdx.x & 31;
    const int row  = (int)(((long long)blockIdx.x * blockDim.x + threadIdx.x) >> 5);
    if (row >= M) return;
    float s = 0.f;
    for (int c = lane; c < HDIM; c += 32)
        s += bf2f(rxn[(long long)row * HDIM + c]) * Wro[c];
    for (int off = 16; off > 0; off >>= 1)
        s += __shfl_down(s, off, 32);
    if (lane == 0) out[row] = s + bro[0];
}

// ---------------------------------------------------------------------------
// Host side
// ---------------------------------------------------------------------------
static size_t alignup(size_t x) { return (x + 255) & ~(size_t)255; }

extern "C" void kernel_launch(void* const* d_in, const int* in_sizes, int n_in,
                              void* d_out, int out_size, void* d_ws, size_t ws_size,
                              hipStream_t stream) {
    (void)n_in; (void)ws_size;
    const int NA = in_sizes[0] / 82;   // 100000 atoms
    const int NB = in_sizes[1] / 88;   // 200000 bonds
    const int NM = out_size;           // 2048 molecules

    const float* fatoms_s = (const float*)d_in[0];
    const float* fbonds_s = (const float*)d_in[1];
    const int*   agraph_s = (const int*)  d_in[2];
    const int*   bgraph_s = (const int*)  d_in[3];
    const int*   midx_s   = (const int*)  d_in[4];
    const float* fatoms_t = (const float*)d_in[5];
    const float* fbonds_t = (const float*)d_in[6];
    const int*   agraph_t = (const int*)  d_in[7];
    const int*   bgraph_t = (const int*)  d_in[8];
    const int*   midx_t   = (const int*)  d_in[9];
    const float* Wi_s = (const float*)d_in[10];
    const float* Wh_s = (const float*)d_in[11];
    const float* Wo_s = (const float*)d_in[12];
    const float* bo_s = (const float*)d_in[13];
    const float* Wi_t = (const float*)d_in[14];
    const float* Wh_t = (const float*)d_in[15];
    const float* Wo_t = (const float*)d_in[16];
    const float* bo_t = (const float*)d_in[17];
    const float* Wrh  = (const float*)d_in[18];
    const float* brh  = (const float*)d_in[19];
    const float* Wro  = (const float*)d_in[20];
    const float* bro  = (const float*)d_in[21];

    // ---- workspace carve-up ----
    char* p = (char*)d_ws;
    auto take = [&](size_t bytes) { char* r = p; p += alignup(bytes); return r; };
    unsigned short* wWi   = (unsigned short*)take((size_t)256 * 96  * 2);
    unsigned short* wWh   = (unsigned short*)take((size_t)256 * 256 * 2);
    unsigned short* wWo   = (unsigned short*)take((size_t)256 * 352 * 2);
    unsigned short* wWrh  = (unsigned short*)take((size_t)256 * 256 * 2);
    unsigned short* fbbf  = (unsigned short*)take((size_t)NB * 96   * 2);
    float*          neiin = (float*)         take((size_t)NB * HDIM * 4);
    unsigned short* msg   = (unsigned short*)take((size_t)NB * HDIM * 2);
    unsigned short* nei   = (unsigned short*)take((size_t)NB * HDIM * 2);
    unsigned short* atin  = (unsigned short*)take((size_t)NA * 352  * 2);
    float*          atomh = (float*)         take((size_t)NA * HDIM * 4);
    float*          mols  = (float*)         take((size_t)NM * HDIM * 4);
    float*          molt  = (float*)         take((size_t)NM * HDIM * 4);
    unsigned short* diffb = (unsigned short*)take((size_t)NM * HDIM * 2);
    unsigned short* rxnb  = (unsigned short*)take((size_t)NM * HDIM * 2);

    const dim3 cvtG(4096), cvtB(256);
    const dim3 gB(256);
    const dim3 gBond((NB / 16 + 7) / 8, HDIM / (NTILES * 16));   // bond GEMMs
    const dim3 gAtom((NA / 16 + 7) / 8, HDIM / (NTILES * 16));   // atom GEMM
    const dim3 gMol ((NM / 16 + 7) / 8, HDIM / (NTILES * 16));   // mol GEMM
    const int  bondWaves = (NB * 32 + 255) / 256;
    const int  atomWaves = (NA * 32 + 255) / 256;

    auto run_side = [&](const float* fatoms, const float* fbonds,
                        const int* agraph, const int* bgraph, const int* midx,
                        const float* Wi, const float* Wh, const float* Wo,
                        const float* bo, float* molh) {
        // weights -> bf16 (padded / concatenated layout)
        k_cvt_pad<<<cvtG, cvtB, 0, stream>>>(Wi, 88, 0,  wWi, 96, 0,  256, 88, 96);
        k_cvt_pad<<<cvtG, cvtB, 0, stream>>>(Wh, 256, 0, wWh, 256, 0, 256, 256, 256);
        k_cvt_pad<<<cvtG, cvtB, 0, stream>>>(Wo, 338, 0,  wWo, 352, 0,  256, 82, 96);
        k_cvt_pad<<<cvtG, cvtB, 0, stream>>>(Wo, 338, 82, wWo, 352, 96, 256, 256, 256);
        // fbonds -> bf16 padded [NB, 96]
        k_cvt_pad<<<cvtG, cvtB, 0, stream>>>(fbonds, 88, 0, fbbf, 96, 0, NB, 88, 96);

        // nei_input_h (f32, pre-relu) and msg0 = relu(...) (bf16) in one pass
        k_gemm_wmma<<<gBond, gB, 0, stream>>>(fbbf, 96, wWi, 96, NB, HDIM, 96,
                                              nullptr, nullptr,
                                              neiin, 0, msg, HDIM);
        // DEPTH-1 = 3 message-passing steps
        for (int it = 0; it < 3; ++it) {
            k_gather_sum<<<bondWaves, gB, 0, stream>>>(msg, bgraph, NB, nei, HDIM, 0);
            k_gemm_wmma<<<gBond, gB, 0, stream>>>(nei, HDIM, wWh, HDIM, NB, HDIM, HDIM,
                                                  neiin, nullptr,
                                                  nullptr, 0, msg, HDIM);
        }
        // atom_in = [fatoms(pad 96) | gather(msg, agraph)]  -> [NA, 352] bf16
        k_cvt_pad<<<cvtG, cvtB, 0, stream>>>(fatoms, 82, 0, atin, 352, 0, NA, 82, 96);
        k_gather_sum<<<atomWaves, gB, 0, stream>>>(msg, agraph, NA, atin, 352, 96);
        // atom_h = relu(atom_in @ Wo_cat^T + bo)  (f32)
        k_gemm_wmma<<<gAtom, gB, 0, stream>>>(atin, 352, wWo, 352, NA, HDIM, 352,
                                              nullptr, bo,
                                              atomh, 1, nullptr, HDIM);
        // mol embed (sorted segment sum, atomic-free)
        k_segsum<<<NM, gB, 0, stream>>>(atomh, midx, NA, molh);
    };

    run_side(fatoms_s, fbonds_s, agraph_s, bgraph_s, midx_s, Wi_s, Wh_s, Wo_s, bo_s, mols);
    run_side(fatoms_t, fbonds_t, agraph_t, bgraph_t, midx_t, Wi_t, Wh_t, Wo_t, bo_t, molt);

    // rxn head
    const int nd = NM * HDIM;
    k_diff_bf<<<(nd + 255) / 256, gB, 0, stream>>>(molt, mols, diffb, nd);
    k_cvt_pad<<<cvtG, cvtB, 0, stream>>>(Wrh, 256, 0, wWrh, 256, 0, 256, 256, 256);
    k_gemm_wmma<<<gMol, gB, 0, stream>>>(diffb, HDIM, wWrh, HDIM, NM, HDIM, HDIM,
                                         nullptr, brh,
                                         nullptr, 0, rxnb, HDIM);
    k_final<<<(NM * 32 + 255) / 256, gB, 0, stream>>>(rxnb, Wro, bro, (float*)d_out, NM);
}
